// LSTMPointerNet_24730421690586
// MI455X (gfx1250) — compile-verified, matrix operand order checked by reading, not run on hardware
//
#include <hip/hip_runtime.h>
#include <hip/hip_bf16.h>
#include <math.h>

typedef __attribute__((ext_vector_type(16))) __bf16 v16bf;
typedef __attribute__((ext_vector_type(8)))  float  v8f;

// ---- fast transcendentals (gfx1250 has V_TANH_F32) -------------------------
#if __has_builtin(__builtin_amdgcn_tanhf)
__device__ __forceinline__ float fast_tanh(float x) { return __builtin_amdgcn_tanhf(x); }
#elif __has_builtin(__builtin_amdgcn_tanh_f32)
__device__ __forceinline__ float fast_tanh(float x) { return __builtin_amdgcn_tanh_f32(x); }
#else
__device__ __forceinline__ float fast_tanh(float x) { return tanhf(x); }
#endif

#if __has_builtin(__builtin_amdgcn_exp2f) && __has_builtin(__builtin_amdgcn_rcpf)
__device__ __forceinline__ float fast_sigmoid(float x) {
  return __builtin_amdgcn_rcpf(1.0f + __builtin_amdgcn_exp2f(-1.4426950408889634f * x));
}
__device__ __forceinline__ float fast_exp(float x) {
  return __builtin_amdgcn_exp2f(1.4426950408889634f * x);
}
#else
__device__ __forceinline__ float fast_sigmoid(float x) { return 1.0f / (1.0f + __expf(-x)); }
__device__ __forceinline__ float fast_exp(float x) { return __expf(x); }
#endif

// ---- Tensor Data Mover availability ---------------------------------------
#if defined(__gfx1250__) && __has_builtin(__builtin_amdgcn_tensor_load_to_lds) && \
    __has_builtin(__builtin_amdgcn_s_wait_tensorcnt)
#define USE_TDM 1
#else
#define USE_TDM 0
#endif

#if USE_TDM
typedef __attribute__((ext_vector_type(4))) unsigned int v4u;
typedef __attribute__((ext_vector_type(8))) int          v8i;
typedef __attribute__((ext_vector_type(4))) int          v4i;

// DMA a 64-row x 32-col bf16 tile (row stride = strideElems in global) into LDS.
// LDS rows padded by hardware: pad_interval=3 (16 dwords = one 64B row),
// pad_amount=3 (4 dwords = 8 bf16) -> LDS row stride 40 bf16 == As[64][40].
__device__ __forceinline__ void tdm_load_tile(const __bf16* gptr,
                                              unsigned int ldsOff,
                                              unsigned int strideElems) {
  unsigned long long ga = (unsigned long long)(uintptr_t)gptr;
  v4u g0;
  g0[0] = 1u;                                                   // count=1 (user D#)
  g0[1] = ldsOff;                                               // lds_addr (bytes)
  g0[2] = (unsigned int)ga;                                     // global_addr[31:0]
  g0[3] = (unsigned int)((ga >> 32) & 0x1FFFFFFu) | (2u << 30); // addr[56:32] | type=2
  v8i g1;
  g1[0] = (int)((1u << 16) | (1u << 20) | (3u << 22) | (3u << 25)); // 2B elems, pad 16dw->+4dw
  g1[1] = (int)((strideElems & 0xFFFFu) << 16);                     // tensor_dim0 lo16
  g1[2] = (int)((strideElems >> 16) | (64u << 16));                 // dim0 hi | tensor_dim1=64
  g1[3] = (int)(32u << 16);                                         // dim1 hi | tile_dim0=32
  g1[4] = (int)64u;                                                 // tile_dim1=64, tile_dim2=0
  g1[5] = (int)strideElems;                                         // tensor_dim0_stride lo32
  g1[6] = 0;                                                        // stride hi | dim1_stride lo
  g1[7] = 0;
  v4i z4 = {0, 0, 0, 0};
#if __clang_major__ >= 23
  v8i z8 = {0, 0, 0, 0, 0, 0, 0, 0};
  __builtin_amdgcn_tensor_load_to_lds(g0, g1, z4, z4, z8, 0);
#else
  __builtin_amdgcn_tensor_load_to_lds(g0, g1, z4, z4, 0);
#endif
}
#endif  // USE_TDM

// pack 8 consecutive f32 -> 8 bf16 (one 16B store)
__device__ __forceinline__ uint4 pack8_bf16(const float* __restrict__ p) {
  float4 a = *(const float4*)p;
  float4 b = *(const float4*)(p + 4);
  union { __bf16 h[8]; uint4 q; } u;
  u.h[0] = (__bf16)a.x; u.h[1] = (__bf16)a.y; u.h[2] = (__bf16)a.z; u.h[3] = (__bf16)a.w;
  u.h[4] = (__bf16)b.x; u.h[5] = (__bf16)b.y; u.h[6] = (__bf16)b.z; u.h[7] = (__bf16)b.w;
  return u.q;
}

// CDNA5 16-bit fragment (ISA 7.12.2): lane's 16 K-values = two contiguous
// 8-elem runs at k = j+8*half and k = 16+j+8*half.
__device__ __forceinline__ v16bf frag16_global(const __bf16* __restrict__ rowbase, int hl) {
  union { v16bf v; uint4 q[2]; } u;
  u.q[0] = *(const uint4*)(rowbase + hl * 8);
  u.q[1] = *(const uint4*)(rowbase + 16 + hl * 8);
  return u.v;
}

// f32 -> bf16 bulk convert
__global__ __launch_bounds__(256) void cvt_f32_bf16(const float* __restrict__ in,
                                                    __bf16* __restrict__ out, int n) {
  int i = (blockIdx.x * 256 + threadIdx.x) * 8;
  if (i + 7 < n) *(uint4*)(out + i) = pack8_bf16(in + i);
}

// transpose + convert: out[c*R + r] = (bf16) in[r*C + c]   (R,C multiples of 32)
__global__ __launch_bounds__(256) void tcvt_f32_bf16(const float* __restrict__ in,
                                                     __bf16* __restrict__ out,
                                                     int R, int Cc) {
  __shared__ float t[32][33];
  const int c0 = blockIdx.x * 32, r0 = blockIdx.y * 32;
  const int tx = threadIdx.x & 31, ty = threadIdx.x >> 5;
#pragma unroll
  for (int i = 0; i < 32; i += 8)
    t[ty + i][tx] = in[(size_t)(r0 + ty + i) * Cc + c0 + tx];
  __syncthreads();
#pragma unroll
  for (int i = 0; i < 32; i += 8)
    out[(size_t)(c0 + ty + i) * R + r0 + tx] = (__bf16)t[tx][ty + i];
}

// ---------------------------------------------------------------------------
// GEMM: C[M,N] = A[M,K] @ BT[N,K]^T (+bias0+bias1) (+= C if accumulate)
// A, BT already bf16, k-contiguous. TDM double-buffered tiles; WMMA compute.
// ---------------------------------------------------------------------------
__global__ __launch_bounds__(256) void gemm_bf16_wmma(
    const __bf16* __restrict__ A, const __bf16* __restrict__ BT,
    float* __restrict__ C, int M, int N, int K,
    const float* __restrict__ bias0, const float* __restrict__ bias1,
    int accumulate) {
  (void)M;
  __shared__ __align__(16) __bf16 As[2][64][40];
  __shared__ __align__(16) __bf16 Bs[2][64][40];

  const int tid  = threadIdx.x;
  const int lane = tid & 31;
  const int wid  = tid >> 5;
  const int hl   = lane >> 4;
  const int l16  = lane & 15;
  const int m0   = blockIdx.y * 64;
  const int n0   = blockIdx.x * 64;
  const int nt   = wid & 3;
  const int mt0  = (wid >> 2) * 2;
  const int ncol = n0 + nt * 16 + l16;
  const int arow0 = mt0 * 16 + l16;
  const int brow  = nt * 16 + l16;

  v8f acc0 = {}, acc1 = {};
  if (accumulate) {
#pragma unroll
    for (int r = 0; r < 8; ++r) {
      acc0[r] = C[(size_t)(m0 + mt0 * 16 + r + 8 * hl) * N + ncol];
      acc1[r] = C[(size_t)(m0 + mt0 * 16 + 16 + r + 8 * hl) * N + ncol];
    }
  }
  if (bias0) {
    float bv = bias0[ncol] + (bias1 ? bias1[ncol] : 0.0f);
#pragma unroll
    for (int r = 0; r < 8; ++r) { acc0[r] += bv; acc1[r] += bv; }
  }

  const int cr = tid >> 2, cq = (tid & 3) * 8;  // fallback copy mapping
  (void)cr; (void)cq;

  // stage chunk 0 into buffer 0
#if USE_TDM
  if (wid == 0) {
    tdm_load_tile(A + (size_t)m0 * K, (unsigned int)(uintptr_t)&As[0][0][0], (unsigned int)K);
    tdm_load_tile(BT + (size_t)n0 * K, (unsigned int)(uintptr_t)&Bs[0][0][0], (unsigned int)K);
  }
#else
  *(uint4*)&As[0][cr][cq] = *(const uint4*)(A + (size_t)(m0 + cr) * K + cq);
  *(uint4*)&Bs[0][cr][cq] = *(const uint4*)(BT + (size_t)(n0 + cr) * K + cq);
#endif

  const int nch = K / 32;
  for (int i = 0; i < nch; ++i) {
    const int cur = i & 1, nxt = cur ^ 1;
    const int kn  = (i + 1) * 32;
    __syncthreads();  // buffer 'nxt' free (chunk i-1 consumed)
#if USE_TDM
    if (wid == 0) {
      if (kn < K) {
        tdm_load_tile(A + (size_t)m0 * K + kn, (unsigned int)(uintptr_t)&As[nxt][0][0],
                      (unsigned int)K);
        tdm_load_tile(BT + (size_t)n0 * K + kn, (unsigned int)(uintptr_t)&Bs[nxt][0][0],
                      (unsigned int)K);
        __builtin_amdgcn_s_wait_tensorcnt(2);  // chunk i landed; i+1 in flight
      } else {
        __builtin_amdgcn_s_wait_tensorcnt(0);
      }
    }
#else
    if (kn < K) {
      *(uint4*)&As[nxt][cr][cq] = *(const uint4*)(A + (size_t)(m0 + cr) * K + kn + cq);
      *(uint4*)&Bs[nxt][cr][cq] = *(const uint4*)(BT + (size_t)(n0 + cr) * K + kn + cq);
    }
#endif
    __syncthreads();  // chunk i visible to all waves

    v16bf af0, af1, bfr;
#pragma unroll
    for (int j = 0; j < 8; ++j) {
      af0[j]     = As[cur][arow0][j + 8 * hl];
      af0[j + 8] = As[cur][arow0][16 + j + 8 * hl];
      af1[j]     = As[cur][arow0 + 16][j + 8 * hl];
      af1[j + 8] = As[cur][arow0 + 16][16 + j + 8 * hl];
      bfr[j]     = Bs[cur][brow][j + 8 * hl];
      bfr[j + 8] = Bs[cur][brow][16 + j + 8 * hl];
    }
    acc0 = __builtin_amdgcn_wmma_f32_16x16x32_bf16(false, af0, false, bfr,
                                                   (short)0, acc0, false, false);
    acc1 = __builtin_amdgcn_wmma_f32_16x16x32_bf16(false, af1, false, bfr,
                                                   (short)0, acc1, false, false);
  }

#pragma unroll
  for (int r = 0; r < 8; ++r) {
    C[(size_t)(m0 + mt0 * 16 + r + 8 * hl) * N + ncol] = acc0[r];
    C[(size_t)(m0 + mt0 * 16 + 16 + r + 8 * hl) * N + ncol] = acc1[r];
  }
}

// ---------------------------------------------------------------------------
// Persistent single-workgroup WMMA LSTM (batch 16 == one WMMA M tile).
// ---------------------------------------------------------------------------
__global__ __launch_bounds__(256) void lstm_wmma(
    const float* __restrict__ X1,      // [16, Lq, 2048]
    const __bf16* __restrict__ WhhBf,  // [2048, 512] bf16 ([n][k])
    float* __restrict__ query,         // [16, Lq, 512]
    int Lq) {
  const int H = 512, G4 = 2048, K = 512;
  __shared__ __align__(16) __bf16 hbf[16][520];
  __shared__ float cs[16][512];

  const int tid  = threadIdx.x;
  const int lane = tid & 31;
  const int wid  = tid >> 5;
  const int hl   = lane >> 4;
  const int l16  = lane & 15;
  const int jc0  = wid * 64;

  for (int e = tid; e < 16 * 512; e += 256) {
    hbf[e >> 9][e & 511] = (__bf16)0.0f;
    cs[e >> 9][e & 511]  = 0.0f;
  }
  __syncthreads();

  for (int t = 0; t < Lq; ++t) {
    v8f acc[4][4];
#pragma unroll
    for (int g = 0; g < 4; ++g)
#pragma unroll
      for (int s = 0; s < 4; ++s) {
        const int n = g * H + jc0 + s * 16 + l16;
#pragma unroll
        for (int r = 0; r < 8; ++r)
          acc[g][s][r] = X1[((size_t)(r + 8 * hl) * Lq + t) * G4 + n];
      }

    for (int kc = 0; kc < K; kc += 32) {
      v16bf af;
#pragma unroll
      for (int j = 0; j < 8; ++j) {
        af[j]     = hbf[l16][kc + j + 8 * hl];
        af[j + 8] = hbf[l16][kc + 16 + j + 8 * hl];
      }
#pragma unroll
      for (int g = 0; g < 4; ++g)
#pragma unroll
        for (int s = 0; s < 4; ++s) {
          const int nrow = g * H + jc0 + s * 16 + l16;
          v16bf bfr = frag16_global(WhhBf + (size_t)nrow * K + kc, hl);
          acc[g][s] = __builtin_amdgcn_wmma_f32_16x16x32_bf16(
              false, af, false, bfr, (short)0, acc[g][s], false, false);
        }
    }
    __syncthreads();

#pragma unroll
    for (int s = 0; s < 4; ++s) {
      const int j = jc0 + s * 16 + l16;
#pragma unroll
      for (int r = 0; r < 8; ++r) {
        const int m  = r + 8 * hl;
        const float ig = fast_sigmoid(acc[0][s][r]);
        const float fg = fast_sigmoid(acc[1][s][r]);
        const float gg = fast_tanh(acc[2][s][r]);
        const float og = fast_sigmoid(acc[3][s][r]);
        const float c  = fg * cs[m][j] + ig * gg;
        cs[m][j] = c;
        const float h = og * fast_tanh(c);
        query[((size_t)m * Lq + t) * H + j] = h;
        hbf[m][j] = (__bf16)h;
      }
    }
    __syncthreads();
  }
}

// ---------------------------------------------------------------------------
// Entity branch fused: additive score -> mask -> softmax -> context.
// ---------------------------------------------------------------------------
__global__ __launch_bounds__(256) void entity_attn_kernel(
    const float* __restrict__ side_feat, const float* __restrict__ sq,
    const float* __restrict__ side_v, const int* __restrict__ entity_nums,
    float* __restrict__ entity_ctx, int Lq, int Le, int H) {
  const int bq = blockIdx.x, b = bq / Lq;
  const int tid = threadIdx.x, lane = tid & 31, wid = tid >> 5;
  __shared__ __align__(16) float sqs[512], vs[512];
  __shared__ float attn[512], red[256];

  for (int h = tid; h < H; h += 256) {
    sqs[h] = sq[(size_t)bq * H + h];
    vs[h]  = side_v[h];
  }
  __syncthreads();
  const int nent = entity_nums[b];

  for (int k = wid; k < Le; k += 8) {
    const float* frow = side_feat + ((size_t)b * Le + k) * H;
    float acc = 0.0f;
    for (int h = lane * 4; h < H; h += 128) {
      float4 f = *(const float4*)(frow + h);
      float4 p = *(const float4*)(sqs + h);
      float4 w = *(const float4*)(vs + h);
      acc += w.x * fast_tanh(f.x + p.x);
      acc += w.y * fast_tanh(f.y + p.y);
      acc += w.z * fast_tanh(f.z + p.z);
      acc += w.w * fast_tanh(f.w + p.w);
    }
#pragma unroll
    for (int off = 16; off > 0; off >>= 1) acc += __shfl_down(acc, off, 32);
    if (lane == 0) attn[k] = (k < nent) ? acc : -1e9f;
  }
  __syncthreads();

  const float sc = (tid < Le) ? attn[tid] : -INFINITY;
  red[tid] = sc;
  __syncthreads();
  for (int s = 128; s > 0; s >>= 1) {
    if (tid < s) red[tid] = fmaxf(red[tid], red[tid + s]);
    __syncthreads();
  }
  const float mx = red[0];
  __syncthreads();
  const float ex = (tid < Le) ? fast_exp(sc - mx) : 0.0f;
  red[tid] = ex;
  __syncthreads();
  for (int s = 128; s > 0; s >>= 1) {
    if (tid < s) red[tid] += red[tid + s];
    __syncthreads();
  }
  const float inv = 1.0f / red[0];
  __syncthreads();
  if (tid < Le) attn[tid] = ex * inv;
  __syncthreads();

  for (int h = tid * 2; h < H; h += 512) {
    float ax = 0.0f, ay = 0.0f;
    for (int k = 0; k < Le; ++k) {
      const float a  = attn[k];
      const float2 f = *(const float2*)(side_feat + ((size_t)b * Le + k) * H + h);
      ax += a * f.x; ay += a * f.y;
    }
    float2 o; o.x = ax; o.y = ay;
    *(float2*)(entity_ctx + (size_t)bq * H + h) = o;
  }
}

// ---------------------------------------------------------------------------
// Final masked pointer scores -> out[b,q,k]
// ---------------------------------------------------------------------------
__global__ __launch_bounds__(256) void sent_score_kernel(
    const float* __restrict__ sent_feat, const float* __restrict__ sproj,
    const float* __restrict__ attn_v, const int* __restrict__ sent_nums,
    float* __restrict__ out, int Lq, int Ls, int H) {
  const int bq = blockIdx.x, b = bq / Lq;
  const int tid = threadIdx.x, lane = tid & 31, wid = tid >> 5;
  __shared__ __align__(16) float ps[512], vs[512];

  for (int h = tid; h < H; h += 256) {
    ps[h] = sproj[(size_t)bq * H + h];
    vs[h] = attn_v[h];
  }
  __syncthreads();
  const int nsent = sent_nums[b];

  for (int k = wid; k < Ls; k += 8) {
    const float* frow = sent_feat + ((size_t)b * Ls + k) * H;
    float acc = 0.0f;
    for (int h = lane * 4; h < H; h += 128) {
      float4 f = *(const float4*)(frow + h);
      float4 p = *(const float4*)(ps + h);
      float4 w = *(const float4*)(vs + h);
      acc += w.x * fast_tanh(f.x + p.x);
      acc += w.y * fast_tanh(f.y + p.y);
      acc += w.z * fast_tanh(f.z + p.z);
      acc += w.w * fast_tanh(f.w + p.w);
    }
#pragma unroll
    for (int off = 16; off > 0; off >>= 1) acc += __shfl_down(acc, off, 32);
    if (lane == 0) out[(size_t)bq * Ls + k] = (k < nsent) ? acc : -1e9f;
  }
}

// ---------------------------------------------------------------------------
extern "C" void kernel_launch(void* const* d_in, const int* in_sizes, int n_in,
                              void* d_out, int out_size, void* d_ws, size_t ws_size,
                              hipStream_t stream) {
  (void)n_in; (void)out_size; (void)ws_size;
  const float* sent_mem    = (const float*)d_in[0];
  const float* entity_mem  = (const float*)d_in[1];
  const float* ptr_in      = (const float*)d_in[2];
  const int*   sent_nums   = (const int*)d_in[3];
  const int*   entity_nums = (const int*)d_in[4];
  const float* Wih         = (const float*)d_in[5];
  const float* Whh         = (const float*)d_in[6];
  const float* bih         = (const float*)d_in[7];
  const float* bhh         = (const float*)d_in[8];
  const float* attn_wm     = (const float*)d_in[9];
  const float* attn_wq     = (const float*)d_in[10];
  const float* attn_v      = (const float*)d_in[11];
  const float* side_wm     = (const float*)d_in[12];
  const float* side_wq     = (const float*)d_in[13];
  const float* side_v      = (const float*)d_in[14];
  const float* attn_ws     = (const float*)d_in[15];

  const int B  = in_sizes[3];            // 16
  const int H  = in_sizes[11];           // 512
  const int S  = in_sizes[12] / H;       // 512
  const int G4 = in_sizes[7];            // 2048
  const int D  = in_sizes[5] / G4;       // 512
  const int Lq = in_sizes[2] / (B * D);  // 64
  const int Ls = in_sizes[0] / (B * D);  // 512
  const int Le = in_sizes[1] / (B * S);  // 256

  // ---- workspace: f32 intermediates then bf16 operand copies --------------
  float* ws = (float*)d_ws;
  float* X1        = ws;                               // [B*Lq, 4H]
  float* query     = X1 + (size_t)B * Lq * G4;
  float* side_feat = query + (size_t)B * Lq * H;
  float* sq        = side_feat + (size_t)B * Le * H;
  float* ectx      = sq + (size_t)B * Lq * H;
  float* sproj     = ectx + (size_t)B * Lq * H;
  float* sent_feat = sproj + (size_t)B * Lq * H;

  __bf16* WhhBf  = (__bf16*)(sent_feat + (size_t)B * Ls * H);  // [4H, H]
  __bf16* WihBf  = WhhBf + (size_t)G4 * H;                     // [4H, D]
  __bf16* wmT    = WihBf + (size_t)G4 * D;                     // attn_wm^T [H, D]
  __bf16* wqT    = wmT + (size_t)H * D;                        // attn_wq^T [H, H]
  __bf16* wsT    = wqT + (size_t)H * H;                        // attn_ws^T [H, H]
  __bf16* swmT   = wsT + (size_t)H * H;                        // side_wm^T [H, S]
  __bf16* swqT   = swmT + (size_t)H * S;                       // side_wq^T [H, H]
  __bf16* ptrBf  = swqT + (size_t)H * H;                       // [B*Lq, D]
  __bf16* entBf  = ptrBf + (size_t)B * Lq * D;                 // [B*Le, S]
  __bf16* sentBf = entBf + (size_t)B * Le * S;                 // [B*Ls, D]
  __bf16* qryBf  = sentBf + (size_t)B * Ls * D;                // [B*Lq, H]
  __bf16* ectxBf = qryBf + (size_t)B * Lq * H;                 // [B*Lq, H]

  dim3 blk(256);
  // ---- one-time operand conversion / weight transposition -----------------
  cvt_f32_bf16<<<dim3((G4 * H) / 2048), blk, 0, stream>>>(Whh, WhhBf, G4 * H);
  cvt_f32_bf16<<<dim3((G4 * D) / 2048), blk, 0, stream>>>(Wih, WihBf, G4 * D);
  cvt_f32_bf16<<<dim3((B * Lq * D) / 2048), blk, 0, stream>>>(ptr_in, ptrBf, B * Lq * D);
  cvt_f32_bf16<<<dim3((B * Le * S) / 2048), blk, 0, stream>>>(entity_mem, entBf, B * Le * S);
  cvt_f32_bf16<<<dim3((B * Ls * D) / 2048), blk, 0, stream>>>(sent_mem, sentBf, B * Ls * D);
  tcvt_f32_bf16<<<dim3(H / 32, D / 32), blk, 0, stream>>>(attn_wm, wmT, D, H);
  tcvt_f32_bf16<<<dim3(H / 32, H / 32), blk, 0, stream>>>(attn_wq, wqT, H, H);
  tcvt_f32_bf16<<<dim3(H / 32, H / 32), blk, 0, stream>>>(attn_ws, wsT, H, H);
  tcvt_f32_bf16<<<dim3(H / 32, S / 32), blk, 0, stream>>>(side_wm, swmT, S, H);
  tcvt_f32_bf16<<<dim3(H / 32, H / 32), blk, 0, stream>>>(side_wq, swqT, H, H);

  // X1 = ptr_in @ Wih.T + (bih + bhh)
  gemm_bf16_wmma<<<dim3(G4 / 64, (B * Lq) / 64), blk, 0, stream>>>(
      ptrBf, WihBf, X1, B * Lq, G4, D, bih, bhh, 0);

  // sequential LSTM recurrence
  lstm_wmma<<<1, blk, 0, stream>>>(X1, WhhBf, query, Lq);
  cvt_f32_bf16<<<dim3((B * Lq * H) / 2048), blk, 0, stream>>>(query, qryBf, B * Lq * H);

  // side_feat = entity_mem @ side_wm ; sq = query @ side_wq
  gemm_bf16_wmma<<<dim3(H / 64, (B * Le) / 64), blk, 0, stream>>>(
      entBf, swmT, side_feat, B * Le, H, S, nullptr, nullptr, 0);
  gemm_bf16_wmma<<<dim3(H / 64, (B * Lq) / 64), blk, 0, stream>>>(
      qryBf, swqT, sq, B * Lq, H, H, nullptr, nullptr, 0);

  // entity additive attention (fused score/softmax/context)
  entity_attn_kernel<<<dim3(B * Lq), blk, 0, stream>>>(
      side_feat, sq, side_v, entity_nums, ectx, Lq, Le, H);
  cvt_f32_bf16<<<dim3((B * Lq * H) / 2048), blk, 0, stream>>>(ectx, ectxBf, B * Lq * H);

  // sproj = query @ attn_wq + entity_ctx @ attn_ws
  gemm_bf16_wmma<<<dim3(H / 64, (B * Lq) / 64), blk, 0, stream>>>(
      qryBf, wqT, sproj, B * Lq, H, H, nullptr, nullptr, 0);
  gemm_bf16_wmma<<<dim3(H / 64, (B * Lq) / 64), blk, 0, stream>>>(
      ectxBf, wsT, sproj, B * Lq, H, H, nullptr, nullptr, 1);

  // sent_feat = sent_mem @ attn_wm
  gemm_bf16_wmma<<<dim3(H / 64, (B * Ls) / 64), blk, 0, stream>>>(
      sentBf, wmT, sent_feat, B * Ls, H, D, nullptr, nullptr, 0);

  // final masked pointer scores -> d_out
  sent_score_kernel<<<dim3(B * Lq), blk, 0, stream>>>(
      sent_feat, sproj, attn_v, sent_nums, (float*)d_out, Lq, Ls, H);
}
